// MultiHeadDistanceLayer_84722524881221
// MI455X (gfx1250) — matrix operand order repaired and checked by reading, not compiled, not run on hardware
//
#include <hip/hip_runtime.h>

// Problem constants (match reference)
#define B_  8
#define F_  256
#define L_  2048
#define H_  8
#define D_  32
#define HD_ 256   // H*D

typedef __attribute__((ext_vector_type(16))) _Float16 v16h;
typedef __attribute__((ext_vector_type(8)))  _Float16 v8h;
typedef __attribute__((ext_vector_type(8)))  float    v8f;

union V16 { v16h v; v8h h[2]; };

// ---------------------------------------------------------------------------
// Kernel 0: transpose W [F, HD] (f32) -> Wt [HD, F] (f16) so B-fragments are
// contiguous 16-half runs per lane. Tiny (128 KB each), runs once per launch.
// ---------------------------------------------------------------------------
__global__ __launch_bounds__(256) void transpose_w(const float* __restrict__ W,
                                                   _Float16* __restrict__ Wt) {
  const int f = blockIdx.x;    // 0..F-1
  const int n = threadIdx.x;   // 0..HD-1  (coalesced read along n)
  Wt[(size_t)n * F_ + f] = (_Float16)W[(size_t)f * HD_ + n];
}

// ---------------------------------------------------------------------------
// Kernel 1: fused Q/K projection GEMM with WMMA.
//   x[b,l,f] = inputs[b,f,l]  (transposed through LDS, f32 -> f16)
//   q = x*Wq + bq, k = x*Wk + bk, stored f16 as [B][H][L][D]
// Block = 256 threads (8 waves) = one 16-row M-tile; wave w owns head w
// (columns [32w, 32w+32) => two 16-col N-subtiles for Wq and for Wk).
// K loop: 8 steps of 32 -> 4 WMMAs per step per wave.
// ---------------------------------------------------------------------------
__global__ __launch_bounds__(256) void proj_kernel(
    const float* __restrict__ inp,      // [B,F,L]
    const _Float16* __restrict__ wtq,   // [HD][F] f16
    const _Float16* __restrict__ wtk,   // [HD][F] f16
    const float* __restrict__ bq,
    const float* __restrict__ bk,
    _Float16* __restrict__ qo,          // [B,H,L,D] f16
    _Float16* __restrict__ ko) {        // [B,H,L,D] f16
  __shared__ _Float16 xs[16][264];      // 16 rows x 256 K, padded (528B row = 33*16B)

  const int tile = blockIdx.x;                 // B*L/16 tiles
  const int b    = tile / (L_ / 16);
  const int l0   = (tile % (L_ / 16)) * 16;
  const int t    = threadIdx.x;

  // Cooperative transposing load of the x tile: coalesced along l.
  {
    const int l  = t & 15;
    const int f0 = (t >> 4) * 16;
    const float* src = inp + (size_t)b * F_ * L_ + (size_t)(l0 + l);
    #pragma unroll
    for (int j = 0; j < 16; ++j)
      xs[l][f0 + j] = (_Float16)src[(size_t)(f0 + j) * L_];
  }
  __syncthreads();

  const int lane = t & 31;
  const int wave = t >> 5;       // == head h
  const int row  = lane & 15;
  const int hi   = lane >> 4;
  const int kbA  = hi * 16;      // A-fragment K chunk base (contiguous 16 K per half)
  const int kbB  = hi * 16;      // B-fragment K chunk base (contiguous 16 K per half, same order as A)
  const int h    = wave;
  const int n0   = h * 32 + row; // N column for subtile 0 (lane holds column n)
  const int n1   = n0 + 16;      // subtile 1

  v8f aq0 = {}, aq1 = {}, ak0 = {}, ak1 = {};
  for (int kc = 0; kc < F_; kc += 32) {
    // A fragment: lane = row (l&15); halves [kb, kb+8) and [kb+8, kb+16)
    V16 a;
    a.h[0] = *(const v8h*)&xs[row][kc + kbA];
    a.h[1] = *(const v8h*)&xs[row][kc + kbA + 8];
    // B fragments: lane = column n; 16 contiguous K-halves at kbB
    const v16h bq0v = *(const v16h*)&wtq[(size_t)n0 * F_ + kc + kbB];
    const v16h bq1v = *(const v16h*)&wtq[(size_t)n1 * F_ + kc + kbB];
    const v16h bk0v = *(const v16h*)&wtk[(size_t)n0 * F_ + kc + kbB];
    const v16h bk1v = *(const v16h*)&wtk[(size_t)n1 * F_ + kc + kbB];
    aq0 = __builtin_amdgcn_wmma_f32_16x16x32_f16(false, a.v, false, bq0v, (short)0, aq0, false, false);
    aq1 = __builtin_amdgcn_wmma_f32_16x16x32_f16(false, a.v, false, bq1v, (short)0, aq1, false, false);
    ak0 = __builtin_amdgcn_wmma_f32_16x16x32_f16(false, a.v, false, bk0v, (short)0, ak0, false, false);
    ak1 = __builtin_amdgcn_wmma_f32_16x16x32_f16(false, a.v, false, bk1v, (short)0, ak1, false, false);
  }

  const float biasq0 = bq[n0], biasq1 = bq[n1];
  const float biask0 = bk[n0], biask1 = bk[n1];
  // C/D layout: VGPR r holds row M = r + hi*8, lane column N = lane&15.
  const size_t base = (((size_t)b * H_ + h) * L_ + l0) * D_;
  #pragma unroll
  for (int r = 0; r < 8; ++r) {
    const int m = r + hi * 8;
    _Float16* qp = qo + base + (size_t)m * D_;
    _Float16* kp = ko + base + (size_t)m * D_;
    qp[row]      = (_Float16)(aq0[r] + biasq0);
    qp[row + 16] = (_Float16)(aq1[r] + biasq1);
    kp[row]      = (_Float16)(ak0[r] + biask0);
    kp[row + 16] = (_Float16)(ak1[r] + biask1);
  }
}

// ---------------------------------------------------------------------------
// Kernel 2: streaming attention. One wave per 16-query tile of one (b,h);
// streams 128 key tiles; each tile is ONE v_wmma_f32_16x16x32_f16 (D=32).
// Online softmax with a SINGLE exp per element: exp(min-max) with selects
// (the other rescale factor is exp(0)=1 by construction). Per-lane partials
// merged across the 16 lanes of each C-fragment row with shfl_xor.
// ---------------------------------------------------------------------------
__global__ __launch_bounds__(256) void attn_kernel(
    const _Float16* __restrict__ qo,    // [B,H,L,D]
    const _Float16* __restrict__ ko,    // [B,H,L,D]
    const float* __restrict__ prior_mean,
    const float* __restrict__ prior_std,
    float* __restrict__ out) {          // [B,L,H]
  const int bid  = blockIdx.x;          // B*H*16
  const int b    = bid / (H_ * 16);
  const int rem  = bid % (H_ * 16);
  const int h    = rem / 16;
  const int grp  = rem % 16;
  const int wave = threadIdx.x >> 5;
  const int lane = threadIdx.x & 31;
  const int q0   = (grp * 8 + wave) * 16;
  const int row  = lane & 15;
  const int hi   = lane >> 4;

  const float mean  = prior_mean[0];
  const float stdv  = prior_std[0];
  const float in2   = -0.5f / (stdv * stdv);
  // fold QK scale (D^-0.5) into the prior amplitude: one mul per element
  const float cs    = (0.3989422804014327f / stdv) * 0.17677669529663687f;

  const size_t bhbase = ((size_t)b * H_ + h) * L_ * D_;

  // Q (A) fragment: lane row = l&15, half-chunks at hi*16 and hi*16+8.
  V16 a;
  {
    const _Float16* qp = qo + bhbase + (size_t)(q0 + row) * D_;
    a.h[0] = *(const v8h*)(qp + hi * 16);
    a.h[1] = *(const v8h*)(qp + hi * 16 + 8);
  }

  float m[8], s[8], w[8], fd[8];
  #pragma unroll
  for (int r = 0; r < 8; ++r) {
    m[r] = -1e30f; s[r] = 0.0f; w[r] = 0.0f;
    // dist for key (j=0, col=row) vs query q0+r+hi*8; advanced by +16 per tile
    fd[r] = (float)(row - (q0 + r + hi * 8));
  }

  const int kbB = hi * 16;
  for (int j = 0; j < L_ / 16; ++j) {
    // K^T (B) fragment: lane column = key j*16 + (l&15); its row in k-storage
    // is that key, 16 contiguous d-halves at kbB -> one aligned 32B load.
    const int key = j * 16 + row;
    const _Float16* kp = ko + bhbase + (size_t)key * D_ + kbB;
    const v16h bfrag = *(const v16h*)kp;
    if (j + 1 < L_ / 16) __builtin_prefetch(kp + 16 * D_, 0, 1);  // global_prefetch_b8

    v8f c = {};
    c = __builtin_amdgcn_wmma_f32_16x16x32_f16(false, a.v, false, bfrag, (short)0, c, false, false);

    #pragma unroll
    for (int r = 0; r < 8; ++r) {
      const float dist = fd[r];
      fd[r] = dist + 16.0f;
      const float dd    = dist - mean;
      const float logit = c[r] * (cs * __expf(in2 * dd * dd));
      // single-exp online softmax update
      const float mo = m[r];
      const bool  gt = logit > mo;
      const float mx = gt ? logit : mo;
      const float e  = __expf((gt ? mo : logit) - mx);
      const float sc = gt ? e : 1.0f;    // rescale factor on accumulators
      const float en = gt ? 1.0f : e;    // weight of the new element
      s[r] = s[r] * sc + en;
      w[r] = w[r] * sc + en * dist;
      m[r] = mx;
    }
  }

  // Merge the 16 per-lane partial softmaxes of each row (lanes sharing hi).
  #pragma unroll
  for (int r = 0; r < 8; ++r) {
    float mr = m[r], sr = s[r], wr = w[r];
    #pragma unroll
    for (int off = 1; off < 16; off <<= 1) {
      const float mo = __shfl_xor(mr, off, 32);
      const float so = __shfl_xor(sr, off, 32);
      const float wo = __shfl_xor(wr, off, 32);
      const float mn = fmaxf(mr, mo);
      const float e0 = __expf(mr - mn);
      const float e1 = __expf(mo - mn);
      sr = sr * e0 + so * e1;
      wr = wr * e0 + wo * e1;
      mr = mn;
    }
    if (row == 0) {
      const int qi = q0 + r + hi * 8;
      out[((size_t)b * L_ + qi) * H_ + h] = wr / sr;
    }
  }
}

// ---------------------------------------------------------------------------
extern "C" void kernel_launch(void* const* d_in, const int* in_sizes, int n_in,
                              void* d_out, int out_size, void* d_ws, size_t ws_size,
                              hipStream_t stream) {
  const float* inputs     = (const float*)d_in[0];   // [B,F,L]
  const float* Wq         = (const float*)d_in[1];   // [F,HD]
  const float* bq         = (const float*)d_in[2];   // [HD]
  const float* Wk         = (const float*)d_in[3];   // [F,HD]
  const float* bk         = (const float*)d_in[4];   // [HD]
  const float* prior_mean = (const float*)d_in[5];   // [1]
  const float* prior_std  = (const float*)d_in[6];   // [1]
  float* out = (float*)d_out;                        // [B,L,H] f32

  // Workspace layout: q (8MB) | k (8MB) | Wt_q (128KB) | Wt_k (128KB)
  char* ws = (char*)d_ws;
  _Float16* qo  = (_Float16*)ws;
  _Float16* ko  = (_Float16*)(ws + (size_t)8 * 1024 * 1024);
  _Float16* wtq = (_Float16*)(ws + (size_t)16 * 1024 * 1024);
  _Float16* wtk = (_Float16*)(ws + (size_t)16 * 1024 * 1024 + (size_t)HD_ * F_ * 2);

  transpose_w<<<F_, HD_, 0, stream>>>(Wq, wtq);
  transpose_w<<<F_, HD_, 0, stream>>>(Wk, wtk);
  proj_kernel<<<B_ * L_ / 16, 256, 0, stream>>>(inputs, wtq, wtk, bq, bk, qo, ko);
  attn_kernel<<<B_ * H_ * 16, 256, 0, stream>>>(qo, ko, prior_mean, prior_std, out);
}
